// RecurrentActorCritic_51754355917108
// MI455X (gfx1250) — matrix operand order repaired and checked by reading, not compiled, not run on hardware
//
#include <hip/hip_runtime.h>
#include <hip/hip_bf16.h>

typedef __attribute__((ext_vector_type(16))) _Float16 v16h;
typedef __attribute__((ext_vector_type(8)))  _Float16 v8h;
typedef __attribute__((ext_vector_type(8)))  float    v8f;
typedef __attribute__((ext_vector_type(4)))  float    v4f;

__device__ __forceinline__ float sigmoidf_(float x) { return 1.0f / (1.0f + __expf(-x)); }

// ---------------- transpose + f32->f16: out[n*K + k] = (f16) in[k*N + n] ----------------
__global__ void transpose_cvt_kernel(const float* __restrict__ in, _Float16* __restrict__ out,
                                     int K, int N) {
  __shared__ float tile[32][33];
  const int kb = blockIdx.y * 32;
  const int nb = blockIdx.x * 32;
  const int tx = threadIdx.x;
  for (int i = threadIdx.y; i < 32; i += 8)
    tile[i][tx] = in[(size_t)(kb + i) * N + (nb + tx)];
  __syncthreads();
  for (int i = threadIdx.y; i < 32; i += 8)
    out[(size_t)(nb + i) * K + (kb + tx)] = (_Float16)tile[tx][i];
}

// ---------------- generic WMMA GEMM ----------------
// out[M,N] = A[M,K](f32, row-major) x Bt[N,K](f16, pre-transposed).
// gridDim.z == 1 : out = act( A*B + bias ), f32 store.
// gridDim.z == S : out is partial buffer [S][M][N]; chunk z covers K/S; no bias/act.
// Block: 256 threads = 8 waves (2 M-tiles x 4 N-groups); wave tile = 16 x 64.
__global__ void gemm_wmma_kernel(const float* __restrict__ A, const _Float16* __restrict__ Bt,
                                 float* __restrict__ out, const float* __restrict__ bias,
                                 int M, int N, int K, int act) {
  const int wave = threadIdx.x >> 5;
  const int lane = threadIdx.x & 31;
  const int wm = wave >> 2, wn = wave & 3;
  const int row0 = blockIdx.y * 32 + wm * 16;
  const int col0 = blockIdx.x * 256 + wn * 64;
  const int m  = lane & 15;
  const int hi = (lane >> 4) & 1;

  v8f acc[4] = {};

  const int kChunk = K / gridDim.z;
  const int kBegin = blockIdx.z * kChunk;
  const float* arow = A + (size_t)(row0 + m) * K;

  for (int k = kBegin; k < kBegin + kChunk; k += 32) {
    // A fragment (16x32 f16 layout): lane<16 -> K = k..k+15 contiguous ; lane>=16 -> k+16..k+31
    const int ka = k + hi * 16;
    const v4f* ap = (const v4f*)(arow + ka);
    v4f a0 = ap[0], a1 = ap[1], a2 = ap[2], a3 = ap[3];
    v16h afrag;
#pragma unroll
    for (int i = 0; i < 4; ++i) {
      afrag[i]      = (_Float16)a0[i];
      afrag[4 + i]  = (_Float16)a1[i];
      afrag[8 + i]  = (_Float16)a2[i];
      afrag[12 + i] = (_Float16)a3[i];
    }
    // B fragment (32x16 f16): lane<16 -> K = k..k+15 contiguous ; lane>=16 -> k+16..k+31
    const int kb = k + hi * 16;
#pragma unroll
    for (int j = 0; j < 4; ++j) {
      v16h bfrag = *(const v16h*)(Bt + (size_t)(col0 + j * 16 + m) * K + kb);
      acc[j] = __builtin_amdgcn_wmma_f32_16x16x32_f16(false, afrag, false, bfrag,
                                                      (short)0, acc[j], false, false);
    }
  }

  // C/D layout: vgpr r -> row = row0 + r + 8*hi, col = col0 + 16*j + (lane&15)
  const int rowBase = row0 + hi * 8;
  if (gridDim.z == 1) {
#pragma unroll
    for (int j = 0; j < 4; ++j) {
      const int cj = col0 + j * 16 + m;
      const float bj = bias ? bias[cj] : 0.0f;
#pragma unroll
      for (int r = 0; r < 8; ++r) {
        float v = acc[j][r] + bj;
        if (act) v = tanhf(v);
        out[(size_t)(rowBase + r) * N + cj] = v;
      }
    }
  } else {
    float* po = out + (size_t)blockIdx.z * M * N;
#pragma unroll
    for (int j = 0; j < 4; ++j) {
      const int cj = col0 + j * 16 + m;
#pragma unroll
      for (int r = 0; r < 8; ++r)
        po[(size_t)(rowBase + r) * N + cj] = acc[j][r];
    }
  }
}

// ---------------- persistent LSTM sequence kernel ----------------
// Runs ALL T time steps of one layer in a single launch. Grid must be (H/64=4, B/32=4) = 16 WGs,
// all co-resident; steps are separated by a device-scope counting barrier (one counter per step,
// zeroed before launch). Each wave owns a 16-row x 16-col h-tile across all 4 gates and keeps its
// entire recurrent-weight slice (8 k-steps x 4 gates x v16h = 256 VGPRs) resident in registers
// for the whole sequence. __launch_bounds__(256, 1): only 2 waves/SIMD exist device-wide, so let
// the compiler allocate the large VGPR budget instead of spilling for occupancy it can't use.
// Z = precomputed x@Wi + b, layout [B*T][4H] with row index b*T+t; gate order [i f g o].
// Wht = [4H][H] f16 transposed recurrent weights. h16a/h16b = [B][H] f16 ping-pong (h16a = h_0).
// c = [B][H] f32 in-place cell state. Y = [B][T][H] f32 hidden outputs. bar = T ints, pre-zeroed.
__global__ void __launch_bounds__(256, 1)
lstm_seq_kernel(const float* __restrict__ Z,
                const _Float16* __restrict__ Wht,
                _Float16* __restrict__ h16a,
                _Float16* __restrict__ h16b,
                float* __restrict__ c,
                float* __restrict__ Y,
                int T, int* __restrict__ bar) {
  const int H = 256;
  const int zStride = T * 4 * H;
  const int yStride = T * H;
  const int nBlk = gridDim.x * gridDim.y;

  const int wave = threadIdx.x >> 5;
  const int lane = threadIdx.x & 31;
  const int wm = wave >> 2, wn = wave & 3;
  const int row0 = blockIdx.y * 32 + wm * 16;
  const int col0 = blockIdx.x * 64 + wn * 16;
  const int m  = lane & 15;
  const int hi = (lane >> 4) & 1;
  const int cj = col0 + m;
  const int rowBase = row0 + hi * 8;

  // preload recurrent-weight fragments for all 8 k-steps x 4 gates (stay in VGPRs across t)
  v16h bf[8][4];
#pragma unroll
  for (int kk = 0; kk < 8; ++kk) {
    const int kb = kk * 32 + hi * 16;
#pragma unroll
    for (int g = 0; g < 4; ++g)
      bf[kk][g] = *(const v16h*)(Wht + (size_t)(g * H + col0 + m) * H + kb);
  }

  for (int t = 0; t < T; ++t) {
    const _Float16* hin  = (t & 1) ? h16b : h16a;
    _Float16*       hout = (t & 1) ? h16a : h16b;
    const float*    Zt   = Z + (size_t)t * 4 * H;
    float*          Yt   = Y + (size_t)t * H;

    v8f acc[4];
#pragma unroll
    for (int g = 0; g < 4; ++g)
#pragma unroll
      for (int r = 0; r < 8; ++r)
        acc[g][r] = Zt[(size_t)(rowBase + r) * zStride + g * H + cj];

    const _Float16* hrow = hin + (row0 + m) * H;
#pragma unroll
    for (int kk = 0; kk < 8; ++kk) {
      const int ka = kk * 32 + hi * 16;
      v8h a0 = *(const v8h*)(hrow + ka);
      v8h a1 = *(const v8h*)(hrow + ka + 8);
      v16h afrag;
#pragma unroll
      for (int i = 0; i < 8; ++i) { afrag[i] = a0[i]; afrag[8 + i] = a1[i]; }
#pragma unroll
      for (int g = 0; g < 4; ++g)
        acc[g] = __builtin_amdgcn_wmma_f32_16x16x32_f16(false, afrag, false, bf[kk][g],
                                                        (short)0, acc[g], false, false);
    }

#pragma unroll
    for (int r = 0; r < 8; ++r) {
      const int row = rowBase + r;
      const float zi = acc[0][r], zf = acc[1][r], zg = acc[2][r], zo = acc[3][r];
      const float cold = c[row * H + cj];
      const float cn = sigmoidf_(zf) * cold + sigmoidf_(zi) * tanhf(zg);
      const float hn = sigmoidf_(zo) * tanhf(cn);
      c[row * H + cj] = cn;
      Yt[(size_t)row * yStride + cj] = hn;
      hout[row * H + cj] = (_Float16)hn;
    }

    // ---- device-scope step barrier (16 co-resident WGs) ----
    __threadfence();       // make this thread's h/c stores visible device-wide
    __syncthreads();       // whole block done with step t
    if (threadIdx.x == 0) {
      __hip_atomic_fetch_add(&bar[t], 1, __ATOMIC_RELEASE, __HIP_MEMORY_SCOPE_AGENT);
      while (__hip_atomic_load(&bar[t], __ATOMIC_ACQUIRE, __HIP_MEMORY_SCOPE_AGENT) < nBlk)
        __builtin_amdgcn_s_sleep(1);
      __threadfence();
    }
    __syncthreads();       // whole block may read step-t h
  }
}

// ---------------- small utility kernels ----------------
__global__ void zero_i32_kernel(int* __restrict__ p, int n) {
  int i = blockIdx.x * blockDim.x + threadIdx.x;
  if (i < n) p[i] = 0;
}
__global__ void copy_f32_kernel(const float* __restrict__ in, float* __restrict__ out, int n) {
  int i = blockIdx.x * blockDim.x + threadIdx.x;
  if (i < n) out[i] = in[i];
}
__global__ void cvt_f16_kernel(const float* __restrict__ in, _Float16* __restrict__ out, int n) {
  int i = blockIdx.x * blockDim.x + threadIdx.x;
  if (i < n) out[i] = (_Float16)in[i];
}
__global__ void gather_last_kernel(const float* __restrict__ Y, float* __restrict__ out,
                                   int T, int H, int B) {
  int i = blockIdx.x * blockDim.x + threadIdx.x;
  if (i >= B * H) return;
  int b = i / H, j = i % H;
  out[i] = Y[((size_t)b * T + (T - 1)) * H + j];
}
__global__ void reduce_splitk_kernel(const float* __restrict__ part, const float* __restrict__ bias,
                                     float* __restrict__ out, int MN, int N, int S, int act) {
  int i = blockIdx.x * blockDim.x + threadIdx.x;
  if (i >= MN) return;
  float s = bias[i % N];
  for (int z = 0; z < S; ++z) s += part[(size_t)z * MN + i];
  out[i] = act ? tanhf(s) : s;
}
__global__ void dense_small_kernel(const float* __restrict__ A, const float* __restrict__ Wm,
                                   const float* __restrict__ bias, float* __restrict__ out,
                                   int B, int N, int K) {
  int i = blockIdx.x * blockDim.x + threadIdx.x;
  if (i >= B * N) return;
  int b = i / N, j = i % N;
  float s = bias[j];
  for (int k = 0; k < K; ++k) s += A[(size_t)b * K + k] * Wm[(size_t)k * N + j];
  out[i] = s;
}

extern "C" void kernel_launch(void* const* d_in, const int* in_sizes, int n_in,
                              void* d_out, int out_size, void* d_ws, size_t ws_size,
                              hipStream_t stream) {
  (void)in_sizes; (void)n_in; (void)out_size; (void)ws_size;
  const int B = 128, T = 256, F = 128, H = 256, W = 512, Aout = 8;
  const int G4H = 4 * H;     // 1024
  const int BT  = B * T;     // 32768
  const int TH  = T * H;     // 65536
  const int SPLITK = 64;

  const float* x    = (const float*)d_in[0];
  const float* c0   = (const float*)d_in[1];
  const float* h0   = (const float*)d_in[2];
  const float* c1   = (const float*)d_in[3];
  const float* h1   = (const float*)d_in[4];
  const float* Wi0  = (const float*)d_in[5];
  const float* Wh0  = (const float*)d_in[6];
  const float* b0   = (const float*)d_in[7];
  const float* Wi1  = (const float*)d_in[8];
  const float* Wh1  = (const float*)d_in[9];
  const float* b1   = (const float*)d_in[10];
  const float* Wv0  = (const float*)d_in[11];
  const float* bv0  = (const float*)d_in[12];
  const float* Wv1  = (const float*)d_in[13];
  const float* bv1  = (const float*)d_in[14];
  const float* Wv2  = (const float*)d_in[15];
  const float* bv2  = (const float*)d_in[16];
  const float* Wm0  = (const float*)d_in[17];
  const float* bm0  = (const float*)d_in[18];
  const float* Wm1  = (const float*)d_in[19];
  const float* bm1  = (const float*)d_in[20];
  const float* Wm2  = (const float*)d_in[21];
  const float* bm2  = (const float*)d_in[22];
  const float* lstd = (const float*)d_in[23];

  // d_out layout (floats): mean[1024] | log_std[8] | value[128] | c0n | h0n | c1n | h1n
  float* out       = (float*)d_out;
  float* out_mean  = out;
  float* out_lstd  = out + 1024;
  float* out_value = out + 1032;
  float* out_c0n   = out + 1160;
  float* out_h0n   = out + 1160 + 32768;
  float* out_c1n   = out + 1160 + 2 * 32768;
  float* out_h1n   = out + 1160 + 3 * 32768;

  // workspace carving (~356 MB total)
  size_t off = 0;
  auto carve = [&](size_t bytes) -> void* {
    void* p = (char*)d_ws + off;
    off += (bytes + 255) & ~(size_t)255;
    return p;
  };
  float*    Z    = (float*)carve((size_t)BT * G4H * 4);     // 128 MB (Z0, reused for Z1)
  float*    Y0   = (float*)carve((size_t)BT * H * 4);       // 32 MB
  float*    Y1   = (float*)carve((size_t)BT * H * 4);       // 32 MB
  _Float16* Wi0t = (_Float16*)carve((size_t)G4H * F * 2);
  _Float16* Wh0t = (_Float16*)carve((size_t)G4H * H * 2);
  _Float16* Wi1t = (_Float16*)carve((size_t)G4H * H * 2);
  _Float16* Wh1t = (_Float16*)carve((size_t)G4H * H * 2);
  _Float16* Wv0t = (_Float16*)carve((size_t)W * TH * 2);    // 64 MB
  _Float16* Wm0t = (_Float16*)carve((size_t)W * TH * 2);    // 64 MB
  _Float16* Wv1t = (_Float16*)carve((size_t)W * W * 2);
  _Float16* Wm1t = (_Float16*)carve((size_t)W * W * 2);
  _Float16* h16a = (_Float16*)carve((size_t)B * H * 2);
  _Float16* h16b = (_Float16*)carve((size_t)B * H * 2);
  float*    Cpart = (float*)carve((size_t)SPLITK * B * W * 4); // 16 MB
  float*    Hv0  = (float*)carve((size_t)B * W * 4);
  float*    Hv1  = (float*)carve((size_t)B * W * 4);
  float*    Hm0  = (float*)carve((size_t)B * W * 4);
  float*    Hm1  = (float*)carve((size_t)B * W * 4);
  int*      bar0 = (int*)carve((size_t)T * 4);
  int*      bar1 = (int*)carve((size_t)T * 4);

  const dim3 tb(32, 8);
  // pre-transpose + convert all weights to f16 [N][K]
  transpose_cvt_kernel<<<dim3(G4H / 32, F / 32), tb, 0, stream>>>(Wi0, Wi0t, F, G4H);
  transpose_cvt_kernel<<<dim3(G4H / 32, H / 32), tb, 0, stream>>>(Wh0, Wh0t, H, G4H);
  transpose_cvt_kernel<<<dim3(G4H / 32, H / 32), tb, 0, stream>>>(Wi1, Wi1t, H, G4H);
  transpose_cvt_kernel<<<dim3(G4H / 32, H / 32), tb, 0, stream>>>(Wh1, Wh1t, H, G4H);
  transpose_cvt_kernel<<<dim3(W / 32, TH / 32), tb, 0, stream>>>(Wv0, Wv0t, TH, W);
  transpose_cvt_kernel<<<dim3(W / 32, TH / 32), tb, 0, stream>>>(Wm0, Wm0t, TH, W);
  transpose_cvt_kernel<<<dim3(W / 32, W / 32), tb, 0, stream>>>(Wv1, Wv1t, W, W);
  transpose_cvt_kernel<<<dim3(W / 32, W / 32), tb, 0, stream>>>(Wm1, Wm1t, W, W);

  copy_f32_kernel<<<1, 8, 0, stream>>>(lstd, out_lstd, 8);
  zero_i32_kernel<<<(2 * T + 255) / 256, 256, 0, stream>>>(bar0, 2 * T); // bar0 and bar1 contiguous

  // ---- layer 0 ----
  copy_f32_kernel<<<(B * H) / 256, 256, 0, stream>>>(c0, out_c0n, B * H); // c lives in d_out slot
  cvt_f16_kernel<<<(B * H) / 256, 256, 0, stream>>>(h0, h16a, B * H);
  gemm_wmma_kernel<<<dim3(G4H / 256, BT / 32, 1), 256, 0, stream>>>(x, Wi0t, Z, b0, BT, G4H, F, 0);
  lstm_seq_kernel<<<dim3(H / 64, B / 32), 256, 0, stream>>>(Z, Wh0t, h16a, h16b, out_c0n, Y0, T, bar0);
  gather_last_kernel<<<(B * H) / 256, 256, 0, stream>>>(Y0, out_h0n, T, H, B);

  // ---- layer 1 ----
  copy_f32_kernel<<<(B * H) / 256, 256, 0, stream>>>(c1, out_c1n, B * H);
  cvt_f16_kernel<<<(B * H) / 256, 256, 0, stream>>>(h1, h16a, B * H);
  gemm_wmma_kernel<<<dim3(G4H / 256, BT / 32, 1), 256, 0, stream>>>(Y0, Wi1t, Z, b1, BT, G4H, H, 0);
  lstm_seq_kernel<<<dim3(H / 64, B / 32), 256, 0, stream>>>(Z, Wh1t, h16a, h16b, out_c1n, Y1, T, bar1);
  gather_last_kernel<<<(B * H) / 256, 256, 0, stream>>>(Y1, out_h1n, T, H, B);

  // ---- value head: flat = Y1 viewed as (B, T*H) ----
  gemm_wmma_kernel<<<dim3(W / 256, B / 32, SPLITK), 256, 0, stream>>>(Y1, Wv0t, Cpart, nullptr, B, W, TH, 0);
  reduce_splitk_kernel<<<(B * W) / 256, 256, 0, stream>>>(Cpart, bv0, Hv0, B * W, W, SPLITK, 1);
  gemm_wmma_kernel<<<dim3(W / 256, B / 32, 1), 256, 0, stream>>>(Hv0, Wv1t, Hv1, bv1, B, W, W, 1);
  dense_small_kernel<<<1, 128, 0, stream>>>(Hv1, Wv2, bv2, out_value, B, 1, W);

  // ---- mean head ----
  gemm_wmma_kernel<<<dim3(W / 256, B / 32, SPLITK), 256, 0, stream>>>(Y1, Wm0t, Cpart, nullptr, B, W, TH, 0);
  reduce_splitk_kernel<<<(B * W) / 256, 256, 0, stream>>>(Cpart, bm0, Hm0, B * W, W, SPLITK, 1);
  gemm_wmma_kernel<<<dim3(W / 256, B / 32, 1), 256, 0, stream>>>(Hm0, Wm1t, Hm1, bm1, B, W, W, 1);
  dense_small_kernel<<<(B * Aout) / 256, 256, 0, stream>>>(Hm1, Wm2, bm2, out_mean, B, Aout, W);
}
